// HFMiMoV2DecoderLayer_61546881352286
// MI455X (gfx1250) — matrix operand order repaired and checked by reading, not compiled, not run on hardware
//
#include <hip/hip_runtime.h>
#include <hip/hip_bf16.h>

// ---------------------------------------------------------------------------
// MiMo-V2 decoder layer on gfx1250 (wave32, WMMA bf16 16x16x32, f32 accum).
// 8-wave (256-thread) blocks, 128x64 C tiles, LDS-shared double-buffered B;
// TDM (tensor_load_to_lds + s_wait_tensorcnt) pipeline for the MoE down-proj.
// ---------------------------------------------------------------------------

#define S_LEN 2048
#define H_DIM 2048
#define NH_   16
#define NKV_  4
#define HD_   128
#define RD_   64
#define E_    16
#define FF_   512
#define MBLK  128           // C-tile rows per block (8 waves x 16)

// 64 rows x 32 bf16 per stage tile; rows padded to 40 elems (80B) so the
// 16-lane B-fragment reads hit 16 distinct LDS banks (stride = 20 words).
#define BROWS  64
#define BPAD   40
#define BSTAGE (BROWS * BPAD)   // elems per buffer (5120 B)

typedef __attribute__((ext_vector_type(16))) __bf16 v16bf;
typedef __attribute__((ext_vector_type(8)))  __bf16 v8bf;
typedef __attribute__((ext_vector_type(8)))  float  v8f;

union AFrag { v16bf v; v8bf h[2]; };

#if defined(__has_builtin)
# if __has_builtin(__builtin_amdgcn_tensor_load_to_lds) && __has_builtin(__builtin_amdgcn_s_wait_tensorcnt)
#  define HAVE_TDM 1
# endif
#endif
#ifndef HAVE_TDM
# define HAVE_TDM 0
#endif

// A fragment (16x32 bf16): lane holds row m=lane&15; one contiguous 16-elem
// run at k = half*16 (matches the contiguous B fragment order).
__device__ __forceinline__ v16bf load_afrag(const __bf16* rowp, int kk, int hf) {
  AFrag f;
  f.h[0] = *(const v8bf*)(rowp + kk + hf * 16);
  f.h[1] = *(const v8bf*)(rowp + kk + hf * 16 + 8);
  return f.v;
}

// ---------------- 8-wave (256-thread) LDS-staged GEMM core ------------------
// Block computes a 128x64 C tile: wave w owns rows [m0blk+16w, +16), all waves
// share the 64-col B tile staged in LDS (double buffered; each thread stages
// one 16B segment, next tile's global load overlaps current tile's WMMAs).
__device__ __forceinline__ void gemm_block128(const __bf16* __restrict__ A,
                                              const __bf16* __restrict__ Bt,
                                              int m0blk, int n0, int K, int ldA, int ldB,
                                              __bf16* lds, v8f acc[4]) {
  const int tid  = threadIdx.x;
  const int wave = tid >> 5, lane = tid & 31;
  const int hf = lane >> 4, lr = lane & 15;
  const __bf16* arow = A + (size_t)(m0blk + wave * 16 + lr) * ldA;
  const int srow = tid >> 2, sseg = tid & 3;          // 4 threads per B row
  const __bf16* gsrc = Bt + (size_t)(n0 + srow) * ldB + sseg * 8;
  const int soff = srow * BPAD + sseg * 8;
#pragma unroll
  for (int t = 0; t < 4; ++t)
#pragma unroll
    for (int r = 0; r < 8; ++r) acc[t][r] = 0.0f;

  v8bf rg = *(const v8bf*)(gsrc);
  *(v8bf*)(lds + soff) = rg;
  __syncthreads();

  int buf = 0;
  for (int kk = 0; kk < K; kk += 32, buf ^= 1) {
    const bool more = (kk + 32) < K;
    if (more) {                                        // issue next-tile load early
      rg = *(const v8bf*)(gsrc + kk + 32);
      if (kk + 256 < K) __builtin_prefetch(gsrc + kk + 256, 0, 1);
    }
    v16bf a = load_afrag(arow, kk, hf);
    const __bf16* bb = lds + buf * BSTAGE + hf * 16;
    v16bf b0 = *(const v16bf*)(bb);                    // preload all fragments so
    v16bf b1 = *(const v16bf*)(bb + 16 * BPAD);        // the backend can use
    v16bf b2 = *(const v16bf*)(bb + 32 * BPAD);        // partial dscnt waits
    v16bf b3 = *(const v16bf*)(bb + 48 * BPAD);
    const __bf16* bbl = bb + lr * BPAD;
    b0 = *(const v16bf*)(bbl);
    b1 = *(const v16bf*)(bbl + 16 * BPAD);
    b2 = *(const v16bf*)(bbl + 32 * BPAD);
    b3 = *(const v16bf*)(bbl + 48 * BPAD);
    acc[0] = __builtin_amdgcn_wmma_f32_16x16x32_bf16(false, a, false, b0, (short)0, acc[0], false, false);
    acc[1] = __builtin_amdgcn_wmma_f32_16x16x32_bf16(false, a, false, b1, (short)0, acc[1], false, false);
    acc[2] = __builtin_amdgcn_wmma_f32_16x16x32_bf16(false, a, false, b2, (short)0, acc[2], false, false);
    acc[3] = __builtin_amdgcn_wmma_f32_16x16x32_bf16(false, a, false, b3, (short)0, acc[3], false, false);
    if (more) *(v8bf*)(lds + (buf ^ 1) * BSTAGE + soff) = rg;
    __syncthreads();
  }
}

// ----------------------- TDM staging (moe_down only) ------------------------
#if HAVE_TDM
typedef unsigned int u32x4 __attribute__((ext_vector_type(4)));
typedef int          i32x8 __attribute__((ext_vector_type(8)));
typedef int          i32x4 __attribute__((ext_vector_type(4)));
typedef __attribute__((address_space(3))) const void lds_cvoid;

__device__ __forceinline__ unsigned int lds_addr32(const void* p) {
  return (unsigned int)(unsigned long long)(lds_cvoid*)p;
}

// Issue a TDM load of a 64-row x 64-byte tile (row stride ldBytes) into LDS.
// data_size = 1 byte; pad: 16 DWORDs data then +4 DWORDs pad -> 80B LDS rows,
// matching the padded compute layout above.
__device__ __forceinline__ void tdm_load_tile(const void* gsrc, unsigned int ldsOff,
                                              unsigned int ldBytes, unsigned int tensorRows) {
  unsigned long long ga = (unsigned long long)gsrc;
  u32x4 g0;
  g0[0] = 1u;                                            // count=1, user descriptor
  g0[1] = ldsOff;                                        // lds_addr
  g0[2] = (unsigned int)ga;                              // global_addr[31:0]
  g0[3] = ((unsigned int)(ga >> 32) & 0x01FFFFFFu) | 0x80000000u;  // addr[56:32] | type=2
  i32x8 g1;
  g1[0] = (int)((1u << 20) | (3u << 22) | (3u << 25));   // pad_en, interval=16DW, amount=4DW
  g1[1] = (int)((ldBytes & 0xFFFFu) << 16);              // tensor_dim0[15:0]
  g1[2] = (int)(((ldBytes >> 16) & 0xFFFFu) | ((tensorRows & 0xFFFFu) << 16));
  g1[3] = (int)(((tensorRows >> 16) & 0xFFFFu) | (64u << 16));   // tile_dim0 = 64 B
  g1[4] = (int)64;                                       // tile_dim1 = 64 rows
  g1[5] = (int)ldBytes;                                  // tensor_dim0_stride[31:0]
  g1[6] = 0;                                             // stride[47:32], dim1_stride=0
  g1[7] = 0;
  i32x4 gz;  gz[0] = 0; gz[1] = 0; gz[2] = 0; gz[3] = 0;
  i32x8 gz8; gz8[0] = 0; gz8[1] = 0; gz8[2] = 0; gz8[3] = 0;
  gz8[4] = 0; gz8[5] = 0; gz8[6] = 0; gz8[7] = 0;
  __builtin_amdgcn_tensor_load_to_lds(g0, g1, gz, gz, gz8, 0);
}
#endif

// ------------------------------- kernels -----------------------------------

__global__ void rmsnorm_kernel(const float* __restrict__ X, const float* __restrict__ W,
                               __bf16* __restrict__ O) {
  __shared__ float red[256];
  const int row = blockIdx.x, tid = threadIdx.x;
  const float* xr = X + (size_t)row * H_DIM;
  float ss = 0.f;
  for (int j = tid; j < H_DIM; j += 256) { float v = xr[j]; ss += v * v; }
  red[tid] = ss; __syncthreads();
  for (int s = 128; s > 0; s >>= 1) { if (tid < s) red[tid] += red[tid + s]; __syncthreads(); }
  const float scale = rsqrtf(red[0] / (float)H_DIM + 1e-6f);
  __bf16* orow = O + (size_t)row * H_DIM;
  for (int j = tid; j < H_DIM; j += 256) orow[j] = (__bf16)(xr[j] * scale * W[j]);
}

// out[n][k] = (bf16) in[k][n];  in is [K][N] f32 row-major. Batched over z.
__global__ void transpose_f32_to_bf16(const float* __restrict__ in, __bf16* __restrict__ out,
                                      int K, int N, size_t inStrideZ, size_t outStrideZ) {
  __shared__ float tile[32][33];
  in  += (size_t)blockIdx.z * inStrideZ;
  out += (size_t)blockIdx.z * outStrideZ;
  const int kb = blockIdx.y * 32, nb = blockIdx.x * 32;
#pragma unroll
  for (int i = 0; i < 4; ++i) {
    int k = kb + threadIdx.y + i * 8;
    tile[threadIdx.y + i * 8][threadIdx.x] = in[(size_t)k * N + nb + threadIdx.x];
  }
  __syncthreads();
#pragma unroll
  for (int i = 0; i < 4; ++i) {
    int n = nb + threadIdx.y + i * 8;
    out[(size_t)n * K + kb + threadIdx.x] = (__bf16)tile[threadIdx.x][threadIdx.y + i * 8];
  }
}

// C = A@B with partial-RoPE epilogue, bf16 store into [head][S][HD].
__global__ void gemm_rope_qk_kernel(const __bf16* __restrict__ A, const __bf16* __restrict__ Bt,
                                    const float* __restrict__ cosb, const float* __restrict__ sinb,
                                    __bf16* __restrict__ Obf, int N, int K) {
  __shared__ __align__(16) __bf16 bst[2 * BSTAGE];
  const int n0 = blockIdx.x * 64;
  v8f acc[4];
  gemm_block128(A, Bt, blockIdx.y * MBLK, n0, K, K, K, bst, acc);
  const int tid = threadIdx.x, wave = tid >> 5, lane = tid & 31;
  const int hf = lane >> 4, lr = lane & 15;
  const int m0 = blockIdx.y * MBLK + wave * 16;
  const int head = n0 / HD_, d0 = n0 % HD_;
  __bf16* obase = Obf + (size_t)head * S_LEN * HD_;
  if (d0 == 0) {  // rotary half: pair d (tile t) with d+32 (tile t+2), same lane
#pragma unroll
    for (int r = 0; r < 8; ++r) {
      int row = m0 + r + 8 * hf;
#pragma unroll
      for (int t = 0; t < 2; ++t) {
        int d = t * 16 + lr;
        float lo = acc[t][r], hi = acc[t + 2][r];
        float cl = cosb[row * RD_ + d],      sl = sinb[row * RD_ + d];
        float ch = cosb[row * RD_ + d + 32], sh = sinb[row * RD_ + d + 32];
        obase[(size_t)row * HD_ + d]      = (__bf16)(lo * cl - hi * sl);
        obase[(size_t)row * HD_ + d + 32] = (__bf16)(hi * ch + lo * sh);
      }
    }
  } else {        // non-rotary half (d in [64,128))
#pragma unroll
    for (int r = 0; r < 8; ++r) {
      int row = m0 + r + 8 * hf;
#pragma unroll
      for (int t = 0; t < 4; ++t)
        obase[(size_t)row * HD_ + 64 + t * 16 + lr] = (__bf16)acc[t][r];
    }
  }
}

// V projection, stored transposed: Vt[kv][d][token] (B-operand layout for PV).
__global__ void gemm_vt_kernel(const __bf16* __restrict__ A, const __bf16* __restrict__ Bt,
                               __bf16* __restrict__ VtO, int N, int K) {
  __shared__ __align__(16) __bf16 bst[2 * BSTAGE];
  const int n0 = blockIdx.x * 64;
  v8f acc[4];
  gemm_block128(A, Bt, blockIdx.y * MBLK, n0, K, K, K, bst, acc);
  const int tid = threadIdx.x, wave = tid >> 5, lane = tid & 31;
  const int hf = lane >> 4, lr = lane & 15;
  const int m0 = blockIdx.y * MBLK + wave * 16;
#pragma unroll
  for (int t = 0; t < 4; ++t) {
    int col = n0 + t * 16 + lr, kvh = col >> 7, d = col & 127;
#pragma unroll
    for (int r = 0; r < 8; ++r) {
      int row = m0 + r + 8 * hf;
      VtO[((size_t)kvh * HD_ + d) * S_LEN + row] = (__bf16)acc[t][r];
    }
  }
}

// C = A@B + res (f32 out).  Wo projection + residual -> h1.
__global__ void gemm_res_kernel(const __bf16* __restrict__ A, const __bf16* __restrict__ Bt,
                                const float* __restrict__ res, float* __restrict__ C,
                                int N, int K) {
  __shared__ __align__(16) __bf16 bst[2 * BSTAGE];
  const int n0 = blockIdx.x * 64;
  v8f acc[4];
  gemm_block128(A, Bt, blockIdx.y * MBLK, n0, K, K, K, bst, acc);
  const int tid = threadIdx.x, wave = tid >> 5, lane = tid & 31;
  const int hf = lane >> 4, lr = lane & 15;
  const int m0 = blockIdx.y * MBLK + wave * 16;
#pragma unroll
  for (int t = 0; t < 4; ++t) {
    int col = n0 + t * 16 + lr;
#pragma unroll
    for (int r = 0; r < 8; ++r) {
      size_t idx = (size_t)(m0 + r + 8 * hf) * N + col;
      C[idx] = acc[t][r] + res[idx];
    }
  }
}

// Flash attention, one wave per (head, 16-query tile). Sink bias in denom.
__global__ void attn_kernel(const __bf16* __restrict__ Q, const __bf16* __restrict__ Kb,
                            const __bf16* __restrict__ Vt, const float* __restrict__ sinkb,
                            __bf16* __restrict__ AO) {
  const int qt = blockIdx.x, h = blockIdx.y;
  const int m0 = qt * 16, kv = h >> 2;   // NH/NKV = 4
  const int lane = threadIdx.x & 31, hf = lane >> 4, lr = lane & 15;
  const float scale = 0.08838834764831845f;  // 1/sqrt(128)

  __shared__ __align__(32) __bf16 pst[16 * 32];

  const __bf16* qrow  = Q + ((size_t)h * S_LEN + (m0 + lr)) * HD_;
  const __bf16* kbase = Kb + (size_t)kv * S_LEN * HD_;
  const __bf16* vbase = Vt + (size_t)kv * HD_ * S_LEN;

  v16bf qf[4];
#pragma unroll
  for (int ds = 0; ds < 4; ++ds) qf[ds] = load_afrag(qrow, ds * 32, hf);

  float mx[8], sm[8];
  v8f o[8];
#pragma unroll
  for (int r = 0; r < 8; ++r) { mx[r] = -1e30f; sm[r] = 0.f; }
#pragma unroll
  for (int nb = 0; nb < 8; ++nb)
#pragma unroll
    for (int r = 0; r < 8; ++r) o[nb][r] = 0.f;

  const int nkt = m0 + 16;
  for (int kt = 0; kt < nkt; kt += 32) {
    v8f s[2];
#pragma unroll
    for (int t2 = 0; t2 < 2; ++t2)
#pragma unroll
      for (int r = 0; r < 8; ++r) s[t2][r] = 0.f;
#pragma unroll
    for (int t2 = 0; t2 < 2; ++t2) {
      if (kt + t2 * 16 < nkt) {
        const __bf16* krow = kbase + (size_t)(kt + t2 * 16 + lr) * HD_ + hf * 16;
#pragma unroll
        for (int ds = 0; ds < 4; ++ds) {
          v16bf b = *(const v16bf*)(krow + ds * 32);
          s[t2] = __builtin_amdgcn_wmma_f32_16x16x32_bf16(false, qf[ds], false, b,
                                                          (short)0, s[t2], false, false);
        }
      }
    }
#pragma unroll
    for (int t2 = 0; t2 < 2; ++t2) {
      int key = kt + t2 * 16 + lr;
      bool live = (kt + t2 * 16 < nkt);
#pragma unroll
      for (int r = 0; r < 8; ++r) {
        int row = m0 + r + 8 * hf;
        s[t2][r] = (live && key <= row) ? s[t2][r] * scale : -1e30f;
      }
    }
#pragma unroll
    for (int r = 0; r < 8; ++r) {
      float rm = fmaxf(s[0][r], s[1][r]);
#pragma unroll
      for (int off = 8; off > 0; off >>= 1) rm = fmaxf(rm, __shfl_xor(rm, off, 16));
      float mnew  = fmaxf(mx[r], rm);
      float alpha = __expf(mx[r] - mnew);
      float p0 = __expf(s[0][r] - mnew);
      float p1 = __expf(s[1][r] - mnew);
      s[0][r] = p0; s[1][r] = p1;
      float rs = p0 + p1;
#pragma unroll
      for (int off = 8; off > 0; off >>= 1) rs += __shfl_xor(rs, off, 16);
      sm[r] = sm[r] * alpha + rs;
      mx[r] = mnew;
#pragma unroll
      for (int nb = 0; nb < 8; ++nb) o[nb][r] *= alpha;
    }
#pragma unroll
    for (int t2 = 0; t2 < 2; ++t2)
#pragma unroll
      for (int r = 0; r < 8; ++r)
        pst[(r + 8 * hf) * 32 + t2 * 16 + lr] = (__bf16)s[t2][r];
    __syncthreads();
    AFrag pf;
    pf.h[0] = *(const v8bf*)(pst + lr * 32 + hf * 16);
    pf.h[1] = *(const v8bf*)(pst + lr * 32 + hf * 16 + 8);
    __syncthreads();
#pragma unroll
    for (int nb = 0; nb < 8; ++nb) {
      v16bf b = *(const v16bf*)(vbase + (size_t)(nb * 16 + lr) * S_LEN + kt + hf * 16);
      o[nb] = __builtin_amdgcn_wmma_f32_16x16x32_bf16(false, pf.v, false, b,
                                                      (short)0, o[nb], false, false);
    }
  }
  float inv[8];
#pragma unroll
  for (int r = 0; r < 8; ++r) inv[r] = 1.0f / (sm[r] + __expf(sinkb[h] - mx[r]));
#pragma unroll
  for (int nb = 0; nb < 8; ++nb)
#pragma unroll
    for (int r = 0; r < 8; ++r) {
      int row = m0 + r + 8 * hf;
      AO[(size_t)row * (NH_ * HD_) + h * HD_ + nb * 16 + lr] = (__bf16)(o[nb][r] * inv[r]);
    }
}

// Sigmoid-gated group-limited top-k router (scalar per token).
__global__ void router_kernel(const __bf16* __restrict__ XF, const float* __restrict__ GW,
                              const float* __restrict__ GB, float* __restrict__ CW) {
  __shared__ float part[E_][8];
  __shared__ float logits[E_];
  const int n = blockIdx.x, tid = threadIdx.x;  // block = 128
  const int e = tid >> 3, sl = tid & 7;
  const __bf16* xr = XF + (size_t)n * H_DIM;
  const float* gw = GW + (size_t)e * H_DIM;
  float s = 0.f;
  for (int j = sl * 256; j < (sl + 1) * 256; ++j) s += (float)xr[j] * gw[j];
  part[e][sl] = s; __syncthreads();
  if (tid < E_) {
    float t = 0.f;
    for (int j = 0; j < 8; ++j) t += part[tid][j];
    logits[tid] = t;
  }
  __syncthreads();
  if (tid == 0) {
    float sg[E_], sfc[E_], tmp[E_], cwl[E_];
    for (int i = 0; i < E_; ++i) {
      sg[i]  = 1.0f / (1.0f + __expf(-logits[i]));
      sfc[i] = sg[i] + GB[i];
      cwl[i] = 0.f;
    }
    float gs[4];
    for (int g = 0; g < 4; ++g) {
      float m1 = -1e30f, m2 = -1e30f;
      for (int j = 0; j < 4; ++j) {
        float v = sfc[g * 4 + j];
        if (v > m1) { m2 = m1; m1 = v; } else if (v > m2) m2 = v;
      }
      gs[g] = m1 + m2;
    }
    int bg0 = 0;
    for (int g = 1; g < 4; ++g) if (gs[g] > gs[bg0]) bg0 = g;
    int bg1 = -1;
    for (int g = 0; g < 4; ++g) if (g != bg0 && (bg1 < 0 || gs[g] > gs[bg1])) bg1 = g;
    for (int i = 0; i < E_; ++i) {
      int grp = i >> 2;
      tmp[i] = (grp == bg0 || grp == bg1) ? sfc[i] : -1e30f;  // masked
    }
    int idx[4]; float tw[4]; float tsum = 0.f;
    for (int i = 0; i < 4; ++i) {
      int be = 0; float bv = -3e30f;
      for (int j = 0; j < E_; ++j) if (tmp[j] > bv) { bv = tmp[j]; be = j; }
      idx[i] = be; tmp[be] = -2e30f;   // removed (still > masked floor)
      tw[i] = sg[be]; tsum += tw[i];
    }
    float invs = 2.5f / (tsum + 1e-20f);
    for (int i = 0; i < 4; ++i) cwl[idx[i]] = tw[i] * invs;
    for (int i = 0; i < E_; ++i) CW[(size_t)n * E_ + i] = cwl[i];
  }
}

// Fused gate+up expert GEMM (128 tok x 64 ff block, both B tiles LDS-shared):
// hgu[tok][e*FF+f] = silu(g)*u*cw[tok][e]  (bf16).
__global__ void moe_gateup_kernel(const __bf16* __restrict__ XF, const __bf16* __restrict__ WegT,
                                  const __bf16* __restrict__ WeuT, const float* __restrict__ CW,
                                  __bf16* __restrict__ HGU) {
  __shared__ __align__(16) __bf16 bstG[2 * BSTAGE];
  __shared__ __align__(16) __bf16 bstU[2 * BSTAGE];
  __shared__ int anyflag;
  const int e     = blockIdx.z;
  const int m0blk = blockIdx.y * MBLK;
  const int n0    = blockIdx.x * 64;
  const int tid = threadIdx.x, wave = tid >> 5, lane = tid & 31;
  const int hf = lane >> 4, lr = lane & 15;
  const int m0 = m0blk + wave * 16;

  if (tid == 0) anyflag = 0;
  __syncthreads();
  if (tid < MBLK && CW[(size_t)(m0blk + tid) * E_ + e] != 0.0f) anyflag = 1;
  __syncthreads();
  if (!anyflag) {  // no token in this 128-row block routes to expert e
#pragma unroll
    for (int t = 0; t < 4; ++t)
#pragma unroll
      for (int r = 0; r < 8; ++r)
        HGU[(size_t)(m0 + r + 8 * hf) * (E_ * FF_) + e * FF_ + n0 + t * 16 + lr] = (__bf16)0.0f;
    return;
  }

  const __bf16* gB = WegT + (size_t)e * FF_ * H_DIM;
  const __bf16* uB = WeuT + (size_t)e * FF_ * H_DIM;
  const __bf16* arow = XF + (size_t)(m0 + lr) * H_DIM;
  const int srow = tid >> 2, sseg = tid & 3;
  const __bf16* gsrcG = gB + (size_t)(n0 + srow) * H_DIM + sseg * 8;
  const __bf16* gsrcU = uB + (size_t)(n0 + srow) * H_DIM + sseg * 8;
  const int soff = srow * BPAD + sseg * 8;

  v8f ag[4], au[4];
#pragma unroll
  for (int t = 0; t < 4; ++t)
#pragma unroll
    for (int r = 0; r < 8; ++r) { ag[t][r] = 0.f; au[t][r] = 0.f; }

  v8bf rG = *(const v8bf*)(gsrcG);
  v8bf rU = *(const v8bf*)(gsrcU);
  *(v8bf*)(bstG + soff) = rG;
  *(v8bf*)(bstU + soff) = rU;
  __syncthreads();

  int buf = 0;
  for (int kk = 0; kk < H_DIM; kk += 32, buf ^= 1) {
    const bool more = (kk + 32) < H_DIM;
    if (more) {
      rG = *(const v8bf*)(gsrcG + kk + 32);
      rU = *(const v8bf*)(gsrcU + kk + 32);
    }
    v16bf a = load_afrag(arow, kk, hf);
    const __bf16* bg = bstG + buf * BSTAGE + lr * BPAD + hf * 16;
    const __bf16* bu = bstU + buf * BSTAGE + lr * BPAD + hf * 16;
    v16bf g0 = *(const v16bf*)(bg);
    v16bf g1 = *(const v16bf*)(bg + 16 * BPAD);
    v16bf g2 = *(const v16bf*)(bg + 32 * BPAD);
    v16bf g3 = *(const v16bf*)(bg + 48 * BPAD);
    v16bf u0 = *(const v16bf*)(bu);
    v16bf u1 = *(const v16bf*)(bu + 16 * BPAD);
    v16bf u2 = *(const v16bf*)(bu + 32 * BPAD);
    v16bf u3 = *(const v16bf*)(bu + 48 * BPAD);
    ag[0] = __builtin_amdgcn_wmma_f32_16x16x32_bf16(false, a, false, g0, (short)0, ag[0], false, false);
    ag[1] = __builtin_amdgcn_wmma_f32_16x16x32_bf16(false, a, false, g1, (short)0, ag[1], false, false);
    ag[2] = __builtin_amdgcn_wmma_f32_16x16x32_bf16(false, a, false, g2, (short)0, ag[2], false, false);
    ag[3] = __builtin_amdgcn_wmma_f32_16x16x32_bf16(false, a, false, g3, (short)0, ag[3], false, false);
    au[0] = __builtin_amdgcn_wmma_f32_16x16x32_bf16(false, a, false, u0, (short)0, au[0], false, false);
    au[1] = __builtin_amdgcn_wmma_f32_16x16x32_bf16(false, a, false, u1, (short)0, au[1], false, false);
    au[2] = __builtin_amdgcn_wmma_f32_16x16x32_bf16(false, a, false, u2, (short)0, au[2], false, false);
    au[3] = __builtin_amdgcn_wmma_f32_16x16x32_bf16(false, a, false, u3, (short)0, au[3], false, false);
    if (more) {
      int o2 = (buf ^ 1) * BSTAGE + soff;
      *(v8bf*)(bstG + o2) = rG;
      *(v8bf*)(bstU + o2) = rU;
    }
    __syncthreads();
  }
#pragma unroll
  for (int t = 0; t < 4; ++t) {
    int f = n0 + t * 16 + lr;
#pragma unroll
    for (int r = 0; r < 8; ++r) {
      int tok = m0 + r + 8 * hf;
      float w = CW[(size_t)tok * E_ + e];
      float g = ag[t][r];
      float val = g * (1.0f / (1.0f + __expf(-g))) * au[t][r] * w;
      HGU[(size_t)tok * (E_ * FF_) + e * FF_ + f] = (__bf16)val;
    }
  }
}

// Down projection (K = E*FF = 8192) + final residual; B tiles staged by the
// Tensor Data Mover when available (double-buffered, s_wait_tensorcnt).
__global__ void moe_down_kernel(const __bf16* __restrict__ HGU, const __bf16* __restrict__ WedT,
                                const float* __restrict__ H1, float* __restrict__ OUT) {
  __shared__ __align__(16) __bf16 bst[2 * BSTAGE];
  const int tid = threadIdx.x, wave = tid >> 5, lane = tid & 31;
  const int hf = lane >> 4, lr = lane & 15;
  const int m0 = blockIdx.y * MBLK + wave * 16;
  const int n0 = blockIdx.x * 64;
  const int K = E_ * FF_;
  v8f acc[4];
#if HAVE_TDM
  const __bf16* arow = HGU + (size_t)(m0 + lr) * K;
  const __bf16* btile = WedT + (size_t)n0 * K;   // 64 rows, stride K elems
#pragma unroll
  for (int t = 0; t < 4; ++t)
#pragma unroll
    for (int r = 0; r < 8; ++r) acc[t][r] = 0.0f;
  if (wave == 0) tdm_load_tile(btile, lds_addr32(bst), (unsigned)K * 2, H_DIM);
  int buf = 0;
  for (int kk = 0; kk < K; kk += 32, buf ^= 1) {
    const bool more = (kk + 32) < K;
    if (wave == 0) {
      if (more) {
        tdm_load_tile((const char*)btile + (kk + 32) * 2,
                      lds_addr32(bst + (buf ^ 1) * BSTAGE), (unsigned)K * 2, H_DIM);
        __builtin_amdgcn_s_wait_tensorcnt(1);   // current tile landed
      } else {
        __builtin_amdgcn_s_wait_tensorcnt(0);
      }
    }
    __syncthreads();
    v16bf a = load_afrag(arow, kk, hf);
    const __bf16* bb = bst + buf * BSTAGE + lr * BPAD + hf * 16;
    v16bf b0 = *(const v16bf*)(bb);
    v16bf b1 = *(const v16bf*)(bb + 16 * BPAD);
    v16bf b2 = *(const v16bf*)(bb + 32 * BPAD);
    v16bf b3 = *(const v16bf*)(bb + 48 * BPAD);
    acc[0] = __builtin_amdgcn_wmma_f32_16x16x32_bf16(false, a, false, b0, (short)0, acc[0], false, false);
    acc[1] = __builtin_amdgcn_wmma_f32_16x16x32_bf16(false, a, false, b1, (short)0, acc[1], false, false);
    acc[2] = __builtin_amdgcn_wmma_f32_16x16x32_bf16(false, a, false, b2, (short)0, acc[2], false, false);
    acc[3] = __builtin_amdgcn_wmma_f32_16x16x32_bf16(false, a, false, b3, (short)0, acc[3], false, false);
    __syncthreads();
  }
#else
  gemm_block128(HGU, WedT, blockIdx.y * MBLK, n0, K, K, K, bst, acc);
#endif
#pragma unroll
  for (int t = 0; t < 4; ++t) {
    int col = n0 + t * 16 + lr;
#pragma unroll
    for (int r = 0; r < 8; ++r) {
      size_t idx = (size_t)(m0 + r + 8 * hf) * H_DIM + col;
      OUT[idx] = acc[t][r] + H1[idx];
    }
  }
}

// ------------------------------- launch ------------------------------------

extern "C" void kernel_launch(void* const* d_in, const int* in_sizes, int n_in,
                              void* d_out, int out_size, void* d_ws, size_t ws_size,
                              hipStream_t stream) {
  const float* x     = (const float*)d_in[0];
  const float* cosb  = (const float*)d_in[1];
  const float* sinb  = (const float*)d_in[2];
  const float* ln1w  = (const float*)d_in[3];
  const float* ln2w  = (const float*)d_in[4];
  const float* Wq    = (const float*)d_in[5];
  const float* Wk    = (const float*)d_in[6];
  const float* Wv    = (const float*)d_in[7];
  const float* Wo    = (const float*)d_in[8];
  const float* sinkb = (const float*)d_in[9];
  const float* gatew = (const float*)d_in[10];
  const float* gateb = (const float*)d_in[11];
  const float* Weg   = (const float*)d_in[12];
  const float* Weu   = (const float*)d_in[13];
  const float* Wed   = (const float*)d_in[14];
  float* outp = (float*)d_out;

  char* base = (char*)d_ws;
  size_t off = 0;
  auto take = [&](size_t bytes) { char* p = base + off; off = (off + bytes + 255) & ~(size_t)255; return p; };
  __bf16* XN   = (__bf16*)take((size_t)S_LEN * H_DIM * 2);
  __bf16* WQT  = (__bf16*)take((size_t)H_DIM * H_DIM * 2);
  __bf16* WKT  = (__bf16*)take((size_t)512 * H_DIM * 2);
  __bf16* WVT  = (__bf16*)take((size_t)512 * H_DIM * 2);
  __bf16* WOT  = (__bf16*)take((size_t)H_DIM * H_DIM * 2);
  __bf16* QB   = (__bf16*)take((size_t)NH_ * S_LEN * HD_ * 2);
  __bf16* KB   = (__bf16*)take((size_t)NKV_ * S_LEN * HD_ * 2);
  __bf16* VTB  = (__bf16*)take((size_t)NKV_ * HD_ * S_LEN * 2);
  __bf16* AOB  = (__bf16*)take((size_t)S_LEN * H_DIM * 2);
  float*  H1   = (float*) take((size_t)S_LEN * H_DIM * 4);
  __bf16* XFB  = (__bf16*)take((size_t)S_LEN * H_DIM * 2);
  float*  CW   = (float*) take((size_t)S_LEN * E_ * 4);
  __bf16* WEGT = (__bf16*)take((size_t)E_ * FF_ * H_DIM * 2);
  __bf16* WEUT = (__bf16*)take((size_t)E_ * FF_ * H_DIM * 2);
  __bf16* WEDT = (__bf16*)take((size_t)H_DIM * E_ * FF_ * 2);
  __bf16* HGU  = (__bf16*)take((size_t)S_LEN * E_ * FF_ * 2);
  (void)ws_size; (void)n_in; (void)in_sizes; (void)out_size;

  const dim3 tb(32, 8);
  rmsnorm_kernel<<<S_LEN, 256, 0, stream>>>(x, ln1w, XN);
  transpose_f32_to_bf16<<<dim3(64, 64, 1),  tb, 0, stream>>>(Wq, WQT, H_DIM, H_DIM, 0, 0);
  transpose_f32_to_bf16<<<dim3(16, 64, 1),  tb, 0, stream>>>(Wk, WKT, H_DIM, 512, 0, 0);
  transpose_f32_to_bf16<<<dim3(16, 64, 1),  tb, 0, stream>>>(Wv, WVT, H_DIM, 512, 0, 0);
  transpose_f32_to_bf16<<<dim3(64, 64, 1),  tb, 0, stream>>>(Wo, WOT, H_DIM, H_DIM, 0, 0);
  transpose_f32_to_bf16<<<dim3(16, 64, E_), tb, 0, stream>>>(Weg, WEGT, H_DIM, FF_,
                                                             (size_t)H_DIM * FF_, (size_t)FF_ * H_DIM);
  transpose_f32_to_bf16<<<dim3(16, 64, E_), tb, 0, stream>>>(Weu, WEUT, H_DIM, FF_,
                                                             (size_t)H_DIM * FF_, (size_t)FF_ * H_DIM);
  transpose_f32_to_bf16<<<dim3(64, 256, 1), tb, 0, stream>>>(Wed, WEDT, E_ * FF_, H_DIM, 0, 0);
  gemm_rope_qk_kernel<<<dim3(32, 16), 256, 0, stream>>>(XN, WQT, cosb, sinb, QB, H_DIM, H_DIM);
  gemm_rope_qk_kernel<<<dim3(8, 16),  256, 0, stream>>>(XN, WKT, cosb, sinb, KB, 512, H_DIM);
  gemm_vt_kernel<<<dim3(8, 16),       256, 0, stream>>>(XN, WVT, VTB, 512, H_DIM);
  attn_kernel<<<dim3(128, NH_), 32, 0, stream>>>(QB, KB, VTB, sinkb, AOB);
  gemm_res_kernel<<<dim3(32, 16), 256, 0, stream>>>(AOB, WOT, x, H1, H_DIM, H_DIM);
  rmsnorm_kernel<<<S_LEN, 256, 0, stream>>>(H1, ln2w, XFB);
  router_kernel<<<S_LEN, 128, 0, stream>>>(XFB, gatew, gateb, CW);
  moe_gateup_kernel<<<dim3(8, 16, E_), 256, 0, stream>>>(XFB, WEGT, WEUT, CW, HGU);
  moe_down_kernel<<<dim3(32, 16), 256, 0, stream>>>(HGU, WEDT, H1, outp);
}
